// TransformerClassifier_6322191859922
// MI455X (gfx1250) — matrix operand order repaired and checked by reading, not compiled, not run on hardware
//
#include <hip/hip_runtime.h>
#include <hip/hip_bf16.h>
#include <cstdint>
#include <cstddef>

#define LAYERS 12
#define BSZ    64
#define SEQ    197
#define SPAD   224
#define CDIM   768
#define NHEAD  12
#define DHEAD  64
#define FFDIM  3072
#define NCLS   1000
#define MROWS  (BSZ * SPAD)          // 14336
#define ATT_SCALE 0.125f             // 1/sqrt(64)

typedef __bf16 bf16_t;
typedef __attribute__((ext_vector_type(16))) __bf16 v16bf;
typedef __attribute__((ext_vector_type(8)))  __bf16 v8bf;
typedef __attribute__((ext_vector_type(8)))  float  v8f;
typedef __attribute__((ext_vector_type(4)))  int    v4i;

// ---------------------------------------------------------------------------
// CDNA5 async Global->LDS DMA (GLOBAL_LOAD_ASYNC_TO_LDS_B128, ASYNCcnt).
// Builtin signature (confirmed by compiler diagnostics): (v4i AS1*, v4i AS3*,
// i32 offset, i32 cpol).
// ---------------------------------------------------------------------------
#if defined(__has_builtin)
#  if __has_builtin(__builtin_amdgcn_global_load_async_to_lds_b128)
#    define HAVE_ASYNC_LDS 1
#  endif
#endif

#if defined(HAVE_ASYNC_LDS)
typedef __attribute__((address_space(1))) v4i gbl_v4i;
typedef __attribute__((address_space(3))) v4i lds_v4i;

static __device__ inline void async_copy_b128(const void* gaddr, void* lds) {
  __builtin_amdgcn_global_load_async_to_lds_b128(
      (gbl_v4i*)gaddr, (lds_v4i*)lds, 0, 0);
}
#  if __has_builtin(__builtin_amdgcn_s_wait_asynccnt)
#    define WAIT_ASYNC(n) __builtin_amdgcn_s_wait_asynccnt(n)
#  else
#    define WAIT_ASYNC(n) asm volatile("s_wait_asynccnt %0" :: "i"(n) : "memory")
#  endif
#endif

// ---------------------------------------------------------------------------
// WMMA fragment helpers (wave32, V_WMMA_F32_16X16X32_BF16)
// A (16x32, M-major LDS tile): lane L -> row m0+(L&15), half h=L>>4,
//   elements 0..15 = K = 16h..16h+15
// B (32x16, stored n-major in LDS): lane L -> col n0+(L&15), half h=L>>4,
//   elements 0..15 = K = 16h..16h+15
// Both are two 16-byte LDS loads per lane.
// ---------------------------------------------------------------------------
static __device__ inline v16bf frag_a(const bf16_t* base, int stride, int m0,
                                      int koff, int lane) {
  const bf16_t* p = base + (size_t)(m0 + (lane & 15)) * stride + koff + 16 * (lane >> 4);
  v8bf lo = *(const v8bf*)p;
  v8bf hi = *(const v8bf*)(p + 8);
  return __builtin_shufflevector(lo, hi, 0,1,2,3,4,5,6,7,8,9,10,11,12,13,14,15);
}

static __device__ inline v16bf frag_b(const bf16_t* base, int stride, int n0,
                                      int koff, int lane) {
  const bf16_t* p = base + (size_t)(n0 + (lane & 15)) * stride + koff + 16 * (lane >> 4);
  v8bf lo = *(const v8bf*)p;
  v8bf hi = *(const v8bf*)(p + 8);
  return __builtin_shufflevector(lo, hi, 0,1,2,3,4,5,6,7,8,9,10,11,12,13,14,15);
}

static __device__ inline v8f wmma_bf16(v16bf a, v16bf b, v8f c) {
  return __builtin_amdgcn_wmma_f32_16x16x32_bf16(false, a, false, b, (short)0, c,
                                                 false, false);
}

static __device__ inline float gelu_exact(float x) {
  return 0.5f * x * (1.0f + erff(x * 0.70710678118654752f));
}

// ---------------------------------------------------------------------------
// Weight conversion with transpose:
//   src fp32 [L][K][N] -> dst bf16 [L][N][K]
// ---------------------------------------------------------------------------
__global__ __launch_bounds__(256) void convert_transpose_kernel(
    const float* __restrict__ src, bf16_t* __restrict__ dst, int K, int N) {
  size_t kn = (size_t)K * N;
  size_t total = (size_t)LAYERS * kn;
  size_t stride = (size_t)gridDim.x * blockDim.x;
  for (size_t i = (size_t)blockIdx.x * blockDim.x + threadIdx.x; i < total;
       i += stride) {
    size_t l = i / kn, rem = i - l * kn;
    int k = (int)(rem / N), n = (int)(rem - (size_t)k * N);
    dst[l * kn + (size_t)n * K + k] = (bf16_t)src[i];
  }
}

// ---------------------------------------------------------------------------
// Embedding: h[b,s,:] = (s==0 ? cls : x[b,s-1]) + pos[s];  pad rows (s>=197)=0
// ---------------------------------------------------------------------------
__global__ __launch_bounds__(256) void embed_kernel(const float* __restrict__ x,
                                                    const float* __restrict__ cls,
                                                    const float* __restrict__ pos,
                                                    float* __restrict__ h) {
  size_t n = (size_t)BSZ * SPAD * CDIM;
  size_t stride = (size_t)gridDim.x * blockDim.x;
  for (size_t i = (size_t)blockIdx.x * blockDim.x + threadIdx.x; i < n; i += stride) {
    int c = (int)(i % CDIM);
    size_t t = i / CDIM;
    int s = (int)(t % SPAD);
    int b = (int)(t / SPAD);
    float v = 0.0f;
    if (s == 0)       v = cls[c] + pos[c];
    else if (s < SEQ) v = x[((size_t)b * (SEQ - 1) + (s - 1)) * CDIM + c] +
                          pos[(size_t)s * CDIM + c];
    h[i] = v;
  }
}

// ---------------------------------------------------------------------------
// Block-wide sum over 256 threads (8 waves, wave32)
// ---------------------------------------------------------------------------
static __device__ inline float block_sum256(float v, float* sred) {
  int tid = threadIdx.x;
#pragma unroll
  for (int o = 16; o > 0; o >>= 1) v += __shfl_down(v, o, 32);
  __syncthreads();
  if ((tid & 31) == 0) sred[tid >> 5] = v;
  __syncthreads();
  float t = 0.0f;
#pragma unroll
  for (int i = 0; i < 8; i++) t += sred[i];
  return t;
}

// ---------------------------------------------------------------------------
// LayerNorm rows of h (fp32, row = one token) -> bf16 y; pad rows zeroed.
// ---------------------------------------------------------------------------
__global__ __launch_bounds__(256) void ln_kernel(const float* __restrict__ h,
                                                 bf16_t* __restrict__ y,
                                                 const float* __restrict__ w,
                                                 const float* __restrict__ b) {
  __shared__ float sred[8];
  int row = blockIdx.x;
  int tid = threadIdx.x;
  const float* xr = h + (size_t)row * CDIM;
  float s = 0.0f;
  for (int j = tid; j < CDIM; j += 256) s += xr[j];
  float mean = block_sum256(s, sred) * (1.0f / CDIM);
  float v = 0.0f;
  for (int j = tid; j < CDIM; j += 256) { float d = xr[j] - mean; v += d * d; }
  float inv = rsqrtf(block_sum256(v, sred) * (1.0f / CDIM) + 1e-5f);
  bool pad = (row % SPAD) >= SEQ;
  bf16_t* yr = y + (size_t)row * CDIM;
  for (int j = tid; j < CDIM; j += 256)
    yr[j] = pad ? (bf16_t)0.0f : (bf16_t)((xr[j] - mean) * inv * w[j] + b[j]);
}

// Final LN of token 0 only (64 rows) -> fp32 y0[64,768]
__global__ __launch_bounds__(256) void final_ln_kernel(const float* __restrict__ h,
                                                       float* __restrict__ y0,
                                                       const float* __restrict__ w,
                                                       const float* __restrict__ b) {
  __shared__ float sred[8];
  int bi = blockIdx.x;
  int tid = threadIdx.x;
  const float* xr = h + (size_t)bi * SPAD * CDIM;  // token 0 of batch bi
  float s = 0.0f;
  for (int j = tid; j < CDIM; j += 256) s += xr[j];
  float mean = block_sum256(s, sred) * (1.0f / CDIM);
  float v = 0.0f;
  for (int j = tid; j < CDIM; j += 256) { float d = xr[j] - mean; v += d * d; }
  float inv = rsqrtf(block_sum256(v, sred) * (1.0f / CDIM) + 1e-5f);
  float* yr = y0 + (size_t)bi * CDIM;
  for (int j = tid; j < CDIM; j += 256)
    yr[j] = (xr[j] - mean) * inv * w[j] + b[j];
}

// ---------------------------------------------------------------------------
// bf16 WMMA GEMM: out = act(A[MxK] * Wt^T + bias), Wt stored [N][K] bf16.
// Block: 256 threads (8 waves), tile 128x128, K step 32.
// Double-buffered async-DMA pipeline: each step issues the next K-tile's
// GLOBAL_LOAD_ASYNC_TO_LDS_B128 into the spare buffer, then waits
// ASYNCcnt<=4 (in-order completion => current tile resident) and runs the
// 8 WMMAs while the DMA engine fills the other buffer.
// ---------------------------------------------------------------------------
template <bool BIAS, bool GELU, bool RESID>
__global__ __launch_bounds__(256) void gemm_kernel(const bf16_t* __restrict__ A,
                                                   const bf16_t* __restrict__ Wt,
                                                   const float* __restrict__ bias,
                                                   bf16_t* __restrict__ outb,
                                                   float* __restrict__ outf,
                                                   int M, int N, int K) {
#if defined(HAVE_ASYNC_LDS)
  __shared__ __align__(16) bf16_t As[2][128 * 40];   // M-major, stride 40
  __shared__ __align__(16) bf16_t Bs[2][128 * 40];   // n-major, stride 40
#else
  __shared__ __align__(16) bf16_t As[1][128 * 40];
  __shared__ __align__(16) bf16_t Bs[1][128 * 40];
#endif

  int tid  = threadIdx.x;
  int lane = tid & 31;
  int w    = tid >> 5;           // wave 0..7
  int wr   = w >> 1;             // wave row 0..3 (32 rows each)
  int wc   = w & 1;              // wave col 0..1 (64 cols each)
  int mBase = blockIdx.y * 128;
  int nBase = blockIdx.x * 128;

  int r0  = tid >> 2;
  int col = (tid & 3) * 8;
  const bf16_t* gA = A  + (size_t)(mBase + r0) * K + col;
  const bf16_t* gB = Wt + (size_t)(nBase + r0) * K + col;
  size_t rowJump = (size_t)64 * K;

  v8f acc[2][4] = {};
  int ksteps = K >> 5;
  int lo = r0 * 40 + col, hi = (r0 + 64) * 40 + col;

#if defined(HAVE_ASYNC_LDS)
  // preload tile 0 via DMA
  async_copy_b128(gA,           &As[0][lo]);
  async_copy_b128(gA + rowJump, &As[0][hi]);
  async_copy_b128(gB,           &Bs[0][lo]);
  async_copy_b128(gB + rowJump, &Bs[0][hi]);
  int buf = 0;
  for (int ks = 0; ks < ksteps; ks++) {
    if (ks + 1 < ksteps) {             // DMA next tile into spare buffer
      gA += 32; gB += 32;
      async_copy_b128(gA,           &As[buf ^ 1][lo]);
      async_copy_b128(gA + rowJump, &As[buf ^ 1][hi]);
      async_copy_b128(gB,           &Bs[buf ^ 1][lo]);
      async_copy_b128(gB + rowJump, &Bs[buf ^ 1][hi]);
      WAIT_ASYNC(4);                   // current tile's 4 transfers are done
    } else {
      WAIT_ASYNC(0);
    }
    __syncthreads();

    v16bf fa0 = frag_a(As[buf], 40, wr * 32, 0, lane);
    v16bf fa1 = frag_a(As[buf], 40, wr * 32 + 16, 0, lane);
#pragma unroll
    for (int nt = 0; nt < 4; nt++) {
      v16bf fb = frag_b(Bs[buf], 40, wc * 64 + nt * 16, 0, lane);
      acc[0][nt] = wmma_bf16(fa0, fb, acc[0][nt]);
      acc[1][nt] = wmma_bf16(fa1, fb, acc[1][nt]);
    }
    __syncthreads();                   // readers done before buf is refilled
    buf ^= 1;
  }
#else
  // Register-staged fallback pipeline
  v8bf a0 = *(const v8bf*)gA;
  v8bf a1 = *(const v8bf*)(gA + rowJump);
  v8bf b0 = *(const v8bf*)gB;
  v8bf b1 = *(const v8bf*)(gB + rowJump);
  for (int ks = 0; ks < ksteps; ks++) {
    *(v8bf*)&As[0][lo] = a0;
    *(v8bf*)&As[0][hi] = a1;
    *(v8bf*)&Bs[0][lo] = b0;
    *(v8bf*)&Bs[0][hi] = b1;
    __syncthreads();
    if (ks + 1 < ksteps) {
      gA += 32; gB += 32;
      a0 = *(const v8bf*)gA;
      a1 = *(const v8bf*)(gA + rowJump);
      b0 = *(const v8bf*)gB;
      b1 = *(const v8bf*)(gB + rowJump);
    }
    v16bf fa0 = frag_a(As[0], 40, wr * 32, 0, lane);
    v16bf fa1 = frag_a(As[0], 40, wr * 32 + 16, 0, lane);
#pragma unroll
    for (int nt = 0; nt < 4; nt++) {
      v16bf fb = frag_b(Bs[0], 40, wc * 64 + nt * 16, 0, lane);
      acc[0][nt] = wmma_bf16(fa0, fb, acc[0][nt]);
      acc[1][nt] = wmma_bf16(fa1, fb, acc[1][nt]);
    }
    __syncthreads();
  }
#endif

  // Epilogue. C layout: VGPR r, lane L -> m = r + 8*(L>>4), n = (L&15)
  int mh = 8 * (lane >> 4);
  int nl = lane & 15;
#pragma unroll
  for (int mt = 0; mt < 2; mt++) {
#pragma unroll
    for (int nt = 0; nt < 4; nt++) {
#pragma unroll
      for (int r = 0; r < 8; r++) {
        int m = mBase + wr * 32 + mt * 16 + r + mh;
        int n = nBase + wc * 64 + nt * 16 + nl;
        float v = acc[mt][nt][r];
        if (BIAS) v += bias[n];
        if (GELU) v = gelu_exact(v);
        if (RESID) outf[(size_t)m * N + n] += v;
        else       outb[(size_t)m * N + n] = (bf16_t)v;
      }
    }
  }
}

// ---------------------------------------------------------------------------
// Attention: one workgroup per (batch, head). S padded to 224.
// qkv layout per token row (2304): [Q(12x64) | K(12x64) | V(12x64)]
// LDS: q[224x72], k[224x72], vT[64x232], probs[224x232]  (~198 KB, dynamic)
// Tiles processed in adjacent pairs (2p, 2p+1): same tile row -> shared
// A-fragment, two independent accumulator chains.
// ---------------------------------------------------------------------------
#define QK_LD  72
#define PV_LD  232
#define ATTN_SMEM_BYTES ((size_t)(SPAD * QK_LD * 2 + DHEAD * PV_LD + SPAD * PV_LD) * sizeof(bf16_t))

__global__ __launch_bounds__(256) void attn_kernel(const bf16_t* __restrict__ qkv,
                                                   bf16_t* __restrict__ o) {
  extern __shared__ __align__(16) bf16_t smem[];
  bf16_t* qs    = smem;                        // [224][72]
  bf16_t* ks    = qs + SPAD * QK_LD;           // [224][72]
  bf16_t* vT    = ks + SPAD * QK_LD;           // [64][232]  (d-major)
  bf16_t* probs = vT + DHEAD * PV_LD;          // [224][232]

  int tid  = threadIdx.x;
  int lane = tid & 31;
  int w    = tid >> 5;
  int b    = blockIdx.x / NHEAD;
  int hh   = blockIdx.x % NHEAD;
  size_t rowbase = (size_t)b * SPAD * (3 * CDIM);

  // Load Q, K (async DMA to LDS when available) and V transposed
  for (int c = tid; c < SPAD * 8; c += 256) {   // 1792 = 7*256: uniform trips
    int s = c >> 3, d8 = (c & 7) * 8;
    size_t g = rowbase + (size_t)s * (3 * CDIM) + hh * DHEAD + d8;
#if defined(HAVE_ASYNC_LDS)
    async_copy_b128(&qkv[g], &qs[s * QK_LD + d8]);
    async_copy_b128(&qkv[g + CDIM], &ks[s * QK_LD + d8]);
#else
    *(v8bf*)&qs[s * QK_LD + d8] = *(const v8bf*)&qkv[g];
    *(v8bf*)&ks[s * QK_LD + d8] = *(const v8bf*)&qkv[g + CDIM];
#endif
  }
  for (int e = tid; e < SPAD * DHEAD; e += 256) {
    int s = e >> 6, d = e & 63;
    vT[d * PV_LD + s] =
        qkv[rowbase + (size_t)s * (3 * CDIM) + 2 * CDIM + hh * DHEAD + d];
  }
#if defined(HAVE_ASYNC_LDS)
  WAIT_ASYNC(0);
#endif
  __syncthreads();

  int mh = 8 * (lane >> 4);
  int nl = lane & 15;

  // Scores: S = (Q K^T) * scale, masked, bf16. 196 tiles = 98 adjacent pairs.
  for (int p = w; p < 98; p += 8) {
    int t0 = 2 * p;
    int tm = t0 / 14, tn0 = t0 % 14, tn1 = tn0 + 1;
    v8f acc0 = {}, acc1 = {};
#pragma unroll
    for (int kc = 0; kc < 2; kc++) {
      v16bf a  = frag_a(qs, QK_LD, tm * 16, kc * 32, lane);
      v16bf b0 = frag_b(ks, QK_LD, tn0 * 16, kc * 32, lane);
      v16bf b1 = frag_b(ks, QK_LD, tn1 * 16, kc * 32, lane);
      acc0 = wmma_bf16(a, b0, acc0);
      acc1 = wmma_bf16(a, b1, acc1);
    }
    int n0 = tn0 * 16 + nl, n1 = tn1 * 16 + nl;
#pragma unroll
    for (int r = 0; r < 8; r++) {
      int m = tm * 16 + r + mh;
      float v0 = acc0[r] * ATT_SCALE;
      float v1 = acc1[r] * ATT_SCALE;
      if (n0 >= SEQ) v0 = -1e30f;
      if (n1 >= SEQ) v1 = -1e30f;
      probs[m * PV_LD + n0] = (bf16_t)v0;
      probs[m * PV_LD + n1] = (bf16_t)v1;
    }
  }
  __syncthreads();

  // Row softmax (one row per thread)
  if (tid < SPAD) {
    bf16_t* row = &probs[tid * PV_LD];
    float mx = -1e30f;
    for (int j = 0; j < SPAD; j++) mx = fmaxf(mx, (float)row[j]);
    float sum = 0.0f;
    for (int j = 0; j < SPAD; j++) {
      float e = __expf((float)row[j] - mx);
      sum += e;
      row[j] = (bf16_t)e;
    }
    float inv = 1.0f / sum;
    for (int j = 0; j < SPAD; j++) row[j] = (bf16_t)((float)row[j] * inv);
  }
  __syncthreads();

  // O = P V : M=224, N=64, K=224 (7 chunks). 56 tiles = 28 adjacent pairs.
  for (int p = w; p < 28; p += 8) {
    int t0 = 2 * p;
    int tm = t0 >> 2, tn0 = t0 & 3, tn1 = tn0 + 1;
    v8f acc0 = {}, acc1 = {};
#pragma unroll
    for (int kc = 0; kc < 7; kc++) {
      v16bf a  = frag_a(probs, PV_LD, tm * 16, kc * 32, lane);
      v16bf b0 = frag_b(vT, PV_LD, tn0 * 16, kc * 32, lane);
      v16bf b1 = frag_b(vT, PV_LD, tn1 * 16, kc * 32, lane);
      acc0 = wmma_bf16(a, b0, acc0);
      acc1 = wmma_bf16(a, b1, acc1);
    }
#pragma unroll
    for (int r = 0; r < 8; r++) {
      int m = tm * 16 + r + mh;
      size_t orow = ((size_t)(b * SPAD + m)) * CDIM + hh * DHEAD;
      o[orow + tn0 * 16 + nl] = (bf16_t)acc0[r];
      o[orow + tn1 * 16 + nl] = (bf16_t)acc1[r];
    }
  }
}

// ---------------------------------------------------------------------------
// Classifier head (tiny: 64x768x1000, fp32 VALU)
// ---------------------------------------------------------------------------
__global__ __launch_bounds__(256) void head_kernel(const float* __restrict__ y0,
                                                   const float* __restrict__ hw,
                                                   const float* __restrict__ hb,
                                                   float* __restrict__ out) {
  int idx = blockIdx.x * 256 + threadIdx.x;
  if (idx >= BSZ * NCLS) return;
  int b = idx / NCLS, n = idx % NCLS;
  const float* yr = y0 + (size_t)b * CDIM;
  float s = hb[n];
  for (int k = 0; k < CDIM; k++) s += yr[k] * hw[(size_t)k * NCLS + n];
  out[idx] = s;
}

// ---------------------------------------------------------------------------
// Launch
// ---------------------------------------------------------------------------
extern "C" void kernel_launch(void* const* d_in, const int* in_sizes, int n_in,
                              void* d_out, int out_size, void* d_ws, size_t ws_size,
                              hipStream_t stream) {
  const float* x       = (const float*)d_in[0];
  const float* cls_emb = (const float*)d_in[1];
  const float* pos_emb = (const float*)d_in[2];
  const float* qkv_w   = (const float*)d_in[3];
  const float* proj_w  = (const float*)d_in[4];
  const float* proj_b  = (const float*)d_in[5];
  const float* ln1_w   = (const float*)d_in[6];
  const float* ln1_b   = (const float*)d_in[7];
  const float* ln2_w   = (const float*)d_in[8];
  const float* ln2_b   = (const float*)d_in[9];
  const float* fc1_w   = (const float*)d_in[10];
  const float* fc1_b   = (const float*)d_in[11];
  const float* fc2_w   = (const float*)d_in[12];
  const float* fc2_b   = (const float*)d_in[13];
  const float* norm_w  = (const float*)d_in[14];
  const float* norm_b  = (const float*)d_in[15];
  const float* head_w  = (const float*)d_in[16];
  const float* head_b  = (const float*)d_in[17];

  char* ws = (char*)d_ws;
  size_t off = 0;
  auto alloc = [&](size_t bytes) -> void* {
    off = (off + 255) & ~(size_t)255;
    void* p = ws + off;
    off += bytes;
    return p;
  };

  const size_t nWQ = (size_t)LAYERS * CDIM * 3 * CDIM;
  const size_t nWP = (size_t)LAYERS * CDIM * CDIM;
  const size_t nW1 = (size_t)LAYERS * CDIM * FFDIM;
  const size_t nW2 = (size_t)LAYERS * FFDIM * CDIM;

  bf16_t* wq   = (bf16_t*)alloc(nWQ * sizeof(bf16_t));  // [L][3C][C]  (N-major)
  bf16_t* wp   = (bf16_t*)alloc(nWP * sizeof(bf16_t));  // [L][C][C]
  bf16_t* w1   = (bf16_t*)alloc(nW1 * sizeof(bf16_t));  // [L][FF][C]
  bf16_t* w2   = (bf16_t*)alloc(nW2 * sizeof(bf16_t));  // [L][C][FF]
  float*  h    = (float*) alloc((size_t)MROWS * CDIM * sizeof(float));
  bf16_t* y    = (bf16_t*)alloc((size_t)MROWS * CDIM * sizeof(bf16_t));
  bf16_t* qkvb = (bf16_t*)alloc((size_t)MROWS * 3 * CDIM * sizeof(bf16_t));
  bf16_t* obuf = (bf16_t*)alloc((size_t)MROWS * CDIM * sizeof(bf16_t));
  bf16_t* ffb  = (bf16_t*)alloc((size_t)MROWS * FFDIM * sizeof(bf16_t));
  float*  y0   = (float*) alloc((size_t)BSZ * CDIM * sizeof(float));

  // Weight conversion fp32 -> bf16, transposed to [N][K] per layer
  convert_transpose_kernel<<<4096, 256, 0, stream>>>(qkv_w, wq, CDIM, 3 * CDIM);
  convert_transpose_kernel<<<4096, 256, 0, stream>>>(proj_w, wp, CDIM, CDIM);
  convert_transpose_kernel<<<4096, 256, 0, stream>>>(fc1_w, w1, CDIM, FFDIM);
  convert_transpose_kernel<<<4096, 256, 0, stream>>>(fc2_w, w2, FFDIM, CDIM);

  // Embedding into fp32 residual stream
  embed_kernel<<<8192, 256, 0, stream>>>(x, cls_emb, pos_emb, h);

  dim3 gQKV(3 * CDIM / 128, MROWS / 128);
  dim3 gPRJ(CDIM / 128, MROWS / 128);
  dim3 gFC1(FFDIM / 128, MROWS / 128);
  dim3 gFC2(CDIM / 128, MROWS / 128);

  for (int l = 0; l < LAYERS; l++) {
    ln_kernel<<<MROWS, 256, 0, stream>>>(h, y, ln1_w + (size_t)l * CDIM,
                                         ln1_b + (size_t)l * CDIM);
    gemm_kernel<false, false, false><<<gQKV, 256, 0, stream>>>(
        y, wq + (size_t)l * CDIM * 3 * CDIM, nullptr, qkvb, nullptr,
        MROWS, 3 * CDIM, CDIM);
    attn_kernel<<<BSZ * NHEAD, 256, ATTN_SMEM_BYTES, stream>>>(qkvb, obuf);
    gemm_kernel<true, false, true><<<gPRJ, 256, 0, stream>>>(
        obuf, wp + (size_t)l * CDIM * CDIM, proj_b + (size_t)l * CDIM,
        nullptr, h, MROWS, CDIM, CDIM);
    ln_kernel<<<MROWS, 256, 0, stream>>>(h, y, ln2_w + (size_t)l * CDIM,
                                         ln2_b + (size_t)l * CDIM);
    gemm_kernel<true, true, false><<<gFC1, 256, 0, stream>>>(
        y, w1 + (size_t)l * CDIM * FFDIM, fc1_b + (size_t)l * FFDIM,
        ffb, nullptr, MROWS, FFDIM, CDIM);
    gemm_kernel<true, false, true><<<gFC2, 256, 0, stream>>>(
        ffb, w2 + (size_t)l * FFDIM * CDIM, fc2_b + (size_t)l * CDIM,
        nullptr, h, MROWS, CDIM, FFDIM);
  }

  final_ln_kernel<<<BSZ, 256, 0, stream>>>(h, y0, norm_w, norm_b);
  head_kernel<<<(BSZ * NCLS + 255) / 256, 256, 0, stream>>>(y0, head_w, head_b,
                                                            (float*)d_out);
}
